// AttentionBlock_67912022884416
// MI455X (gfx1250) — matrix operand order repaired and checked by reading, not compiled, not run on hardware
//
#include <hip/hip_runtime.h>
#include <hip/hip_bf16.h>

// ---------------------------------------------------------------------------
// AttentionBlock: GroupNorm -> QKV (1x1) -> softmax(Q^T K / sqrt(C)) V -> proj
//                 -> residual.   B=2, C=512, H=W=64, N=4096, G=32.
// Compute-bound (~86 GF) -> everything through v_wmma_f32_16x16x32_f16.
// Weights pre-converted to f16 once (removes cvt chains + WMMA hazard nops
// from the hot loops).  Flash-style 2-pass softmax; K-panels staged with
// global_load_async_to_lds_b128 (double-buffered, shared by 4 waves/block).
// ---------------------------------------------------------------------------

#define BATCH 2
#define CDIM 512
#define NPix 4096
#define NGROUPS 32
#define CG (CDIM / NGROUPS)   // 16
#define EPS 1e-6f
#define QK_SCALE 0.044194173824159216f  // 1/sqrt(512)
#define WELEMS (CDIM * CDIM)            // elements per weight matrix

typedef _Float16 half_t;
typedef __attribute__((ext_vector_type(16))) _Float16 v16h;
typedef __attribute__((ext_vector_type(8)))  _Float16 v8h;
typedef __attribute__((ext_vector_type(8)))  float    v8f;
typedef __attribute__((ext_vector_type(4)))  float    v4f;

typedef __attribute__((address_space(3))) const void* lds_cptr_t;

__device__ __forceinline__ v8f zero8() {
    v8f z = {0.f, 0.f, 0.f, 0.f, 0.f, 0.f, 0.f, 0.f};
    return z;
}

// Raw 32-bit LDS byte address of a __shared__ object (for async-LDS DMA).
__device__ __forceinline__ uint32_t lds_addr_of(const void* p) {
    return (uint32_t)(size_t)(lds_cptr_t)p;
}

#define S_WAIT_ASYNCCNT(n) asm volatile("s_wait_asynccnt " #n ::: "memory")
#define S_WAIT_DSCNT0()    asm volatile("s_wait_dscnt 0" ::: "memory")

// Stage one contiguous 16KB panel (16 rows x 512 f16) global -> LDS with the
// CDNA5 async-LDS DMA path (ASYNCcnt).  blockDim.x == 128: each thread issues
// 8 x b128, i.e. 8 async instructions per wave per panel.
__device__ __forceinline__ void stage_panel_async(uint32_t lds_base,
                                                  const half_t* gsrc) {
    const char* g = (const char*)gsrc;
    int tid = threadIdx.x;
#pragma unroll
    for (int q = 0; q < 8; ++q) {
        int off = (q * 128 + tid) * 16;
        asm volatile("global_load_async_to_lds_b128 %0, %1, off"
                     :: "v"(lds_base + (uint32_t)off), "v"(g + off)
                     : "memory");
    }
}

// A-matrix fragment (16x32 f16, MxK) from row-major f16 [row][k], stride ld.
// lane L: row (L&15); halves 0..15 -> K=koff+0..15,
// koff = (L>>4)*16.
__device__ __forceinline__ v16h load_a_f16(const half_t* base, int ld) {
    int lane = threadIdx.x & 31;
    int row  = lane & 15;
    int koff = (lane >> 4) << 4;
    const half_t* p = base + row * ld;
    v8h lo = *(const v8h*)(p + koff);
    v8h hi = *(const v8h*)(p + 8 + koff);
    v16h a;
#pragma unroll
    for (int i = 0; i < 8; ++i) { a[i] = lo[i]; a[8 + i] = hi[i]; }
    return a;
}

// B-matrix fragment (32x16 f16, KxN) from memory laid out src[col][k], ld.
// lane L: column (L&15); halves 0..15 -> K = (L>>4)*16 + 0..15.
__device__ __forceinline__ v16h load_b_kcontig(const half_t* base, int ld) {
    int lane = threadIdx.x & 31;
    int col  = lane & 15;
    int krun = (lane >> 4) << 4;
    return *(const v16h*)(base + col * ld + krun);
}

__device__ __forceinline__ v8f wmma_f16(v16h a, v16h b, v8f c) {
    return __builtin_amdgcn_wmma_f32_16x16x32_f16(
        false, a, false, b, (short)0, c, false, false);
}

// Same WMMA with the reuse-A OPSEL hint: legal because it always follows an
// identical WMMA carrying the same A operand (ISA 7.12 VOP3P RA hint).
__device__ __forceinline__ v8f wmma_f16_ra(v16h a, v16h b, v8f c) {
    return __builtin_amdgcn_wmma_f32_16x16x32_f16(
        false, a, false, b, (short)0, c, true, false);
}

// ---------------------------------------------------------------------------
// K0: one-shot weight conversion f32 -> f16 (wq|wk|wv|wp -> wh[4][512*512]).
// ---------------------------------------------------------------------------
__global__ void wcvt_kernel(const float* __restrict__ wq,
                            const float* __restrict__ wk,
                            const float* __restrict__ wv,
                            const float* __restrict__ wp,
                            half_t* __restrict__ wh) {
    int z = blockIdx.y;
    const float* src = (z == 0) ? wq : (z == 1) ? wk : (z == 2) ? wv : wp;
    half_t* dst = wh + (size_t)z * WELEMS;
    for (int i = blockIdx.x * blockDim.x + threadIdx.x; i < WELEMS / 4;
         i += gridDim.x * blockDim.x) {
        v4f v = *(const v4f*)(src + 4 * i);
        half_t h0 = (half_t)v[0], h1 = (half_t)v[1];
        half_t h2 = (half_t)v[2], h3 = (half_t)v[3];
        dst[4 * i + 0] = h0; dst[4 * i + 1] = h1;
        dst[4 * i + 2] = h2; dst[4 * i + 3] = h3;
    }
}

// ---------------------------------------------------------------------------
// K1: GroupNorm.  One block per (b, group).  Writes hnT[n][c] (n-major f16).
// ---------------------------------------------------------------------------
__global__ void gn_kernel(const float* __restrict__ x,
                          const float* __restrict__ gsc,
                          const float* __restrict__ gbs,
                          half_t* __restrict__ hnT) {
    int b = blockIdx.x / NGROUPS;
    int g = blockIdx.x % NGROUPS;
    const float* xp = x + (size_t)b * CDIM * NPix + (size_t)g * CG * NPix;

    __shared__ float s1[256];
    __shared__ float s2[256];
    float sum = 0.f, ssq = 0.f;
    for (int idx = threadIdx.x; idx < CG * NPix; idx += 256) {
        float v = xp[idx];
        sum += v; ssq += v * v;
    }
    s1[threadIdx.x] = sum; s2[threadIdx.x] = ssq;
    __syncthreads();
    for (int s = 128; s > 0; s >>= 1) {
        if ((int)threadIdx.x < s) {
            s1[threadIdx.x] += s1[threadIdx.x + s];
            s2[threadIdx.x] += s2[threadIdx.x + s];
        }
        __syncthreads();
    }
    const float cnt = (float)(CG * NPix);
    float mean = s1[0] / cnt;
    float var  = s2[0] / cnt - mean * mean;
    float inv  = rsqrtf(var + EPS);

    for (int idx = threadIdx.x; idx < CG * NPix; idx += 256) {
        int cl = idx >> 12;
        int n  = idx & (NPix - 1);
        int c  = g * CG + cl;
        float v = (xp[idx] - mean) * inv * gsc[c] + gbs[c];
        hnT[(size_t)b * NPix * CDIM + (size_t)n * CDIM + c] = (half_t)v;
    }
}

// ---------------------------------------------------------------------------
// Fully-unrolled K-loop GEMM body shared by the projection kernels:
// acc[4] (16d x 64n) = W(16x512 f16) * srcT^T, srcT is [n][c] f16.
// Straight-line: scheduler hoists loads across steps; all-f16 operands mean
// no VALU writes into WMMA sources (no hazard nops).
// ---------------------------------------------------------------------------
__device__ __forceinline__ void gemm_row_panel(const half_t* w, int dbase,
                                               const half_t* srcT, int nbase,
                                               v8f acc[4]) {
#pragma unroll
    for (int k = 0; k < CDIM; k += 32) {
        v16h a  = load_a_f16(w + (size_t)dbase * CDIM + k, CDIM);
        v16h b0 = load_b_kcontig(srcT + (size_t)(nbase + 0)  * CDIM + k, CDIM);
        v16h b1 = load_b_kcontig(srcT + (size_t)(nbase + 16) * CDIM + k, CDIM);
        v16h b2 = load_b_kcontig(srcT + (size_t)(nbase + 32) * CDIM + k, CDIM);
        v16h b3 = load_b_kcontig(srcT + (size_t)(nbase + 48) * CDIM + k, CDIM);
        acc[0] = wmma_f16(a, b0, acc[0]);
        acc[1] = wmma_f16_ra(a, b1, acc[1]);
        acc[2] = wmma_f16_ra(a, b2, acc[2]);
        acc[3] = wmma_f16_ra(a, b3, acc[3]);
    }
}

// ---------------------------------------------------------------------------
// K2: Q/K/V projections.  gridDim.z = b*3 + proj; wave = 16(d) x 64(n) tile.
//     Q -> qT[n][d] (pre-scaled), K -> kT[n][d], V -> v[d][n].
//     Branch-free epilogue via (dst, strideN, strideD, scale).
// ---------------------------------------------------------------------------
__global__ void qkv_kernel(const half_t* __restrict__ hnT,
                           const half_t* __restrict__ wh,
                           const float* __restrict__ bq,
                           const float* __restrict__ bk,
                           const float* __restrict__ bv,
                           half_t* __restrict__ qT, half_t* __restrict__ kT,
                           half_t* __restrict__ vM) {
    int z = blockIdx.z;
    int b = z / 3, proj = z % 3;
    const half_t* w  = wh + (size_t)proj * WELEMS;
    const float*  bb = (proj == 0) ? bq : (proj == 1) ? bk : bv;

    int wave  = threadIdx.x >> 5;
    int dbase = (blockIdx.y * 4 + wave) * 16;
    int nbase = blockIdx.x * 64;
    const half_t* hb = hnT + (size_t)b * NPix * CDIM;

    v8f acc[4] = {zero8(), zero8(), zero8(), zero8()};
    gemm_row_panel(w, dbase, hb, nbase, acc);

    // Epilogue config (uniform per block -> scalar selects, no per-lane branch)
    half_t* dst;
    size_t sN, sD;
    float scale;
    if (proj == 0)      { dst = qT + (size_t)b * NPix * CDIM; sN = CDIM; sD = 1;   scale = QK_SCALE; }
    else if (proj == 1) { dst = kT + (size_t)b * NPix * CDIM; sN = CDIM; sD = 1;   scale = 1.f; }
    else                { dst = vM + (size_t)b * CDIM * NPix; sN = 1;   sD = NPix; scale = 1.f; }

    int lane = threadIdx.x & 31;
    int hsel = lane >> 4, col = lane & 15;
    float bias_r[8];
#pragma unroll
    for (int r = 0; r < 8; ++r) bias_r[r] = bb[dbase + r + hsel * 8];

#pragma unroll
    for (int t = 0; t < 4; ++t) {
#pragma unroll
        for (int r = 0; r < 8; ++r) {
            size_t d = (size_t)(dbase + r + hsel * 8);
            size_t n = (size_t)(nbase + t * 16 + col);
            dst[n * sN + d * sD] = (half_t)((acc[t][r] + bias_r[r]) * scale);
        }
    }
}

// ---------------------------------------------------------------------------
// K3: flash pass 1 (row max m, row sum-of-exp l).  4 waves/block share the
//     K-panel via double-buffered async-LDS staging; Q fragments stay in VGPRs.
// ---------------------------------------------------------------------------
__global__ void stats_kernel(const half_t* __restrict__ qT,
                             const half_t* __restrict__ kT,
                             float* __restrict__ ms, float* __restrict__ ls) {
    int b = blockIdx.y;
    int wave = threadIdx.x >> 5;
    int ibase = (blockIdx.x * 4 + wave) * 16;
    const half_t* qb = qT + (size_t)b * NPix * CDIM;
    const half_t* kb = kT + (size_t)b * NPix * CDIM;

    __shared__ __align__(128) half_t panel[2][16 * CDIM];   // 2 x 16KB
    uint32_t pbase0 = lds_addr_of(&panel[0][0]);
    uint32_t pbase1 = lds_addr_of(&panel[1][0]);

    v16h qf[16];
#pragma unroll
    for (int ks = 0; ks < 16; ++ks)
        qf[ks] = load_a_f16(qb + (size_t)ibase * CDIM + ks * 32, CDIM);

    float m[8], l[8];
#pragma unroll
    for (int r = 0; r < 8; ++r) { m[r] = -1e30f; l[r] = 0.f; }

    stage_panel_async(pbase0, kb);                 // panel h=0
    for (int h = 0; h < NPix / 16; ++h) {
        const half_t* pp = (h & 1) ? &panel[1][0] : &panel[0][0];
        if (h + 1 < NPix / 16) {                   // prefetch next panel
            stage_panel_async((h & 1) ? pbase0 : pbase1,
                              kb + (size_t)(h + 1) * 16 * CDIM);
            S_WAIT_ASYNCCNT(8);                    // only wait for panel h
        } else {
            S_WAIT_ASYNCCNT(0);
        }
        __syncthreads();

        v8f acc = zero8();
#pragma unroll
        for (int ks = 0; ks < 16; ++ks)
            acc = wmma_f16(qf[ks], load_b_kcontig(pp + ks * 32, CDIM), acc);

#pragma unroll
        for (int r = 0; r < 8; ++r) {
            float s  = acc[r];
            float tm = s;
            tm = fmaxf(tm, __shfl_xor(tm, 1, 32));
            tm = fmaxf(tm, __shfl_xor(tm, 2, 32));
            tm = fmaxf(tm, __shfl_xor(tm, 4, 32));
            tm = fmaxf(tm, __shfl_xor(tm, 8, 32));
            float mn = fmaxf(m[r], tm);
            float p  = __expf(s - mn);
            float ts = p;
            ts += __shfl_xor(ts, 1, 32);
            ts += __shfl_xor(ts, 2, 32);
            ts += __shfl_xor(ts, 4, 32);
            ts += __shfl_xor(ts, 8, 32);
            l[r] = l[r] * __expf(m[r] - mn) + ts;
            m[r] = mn;
        }
        __syncthreads();                           // done reading this buffer
    }

    int lane = threadIdx.x & 31;
    if ((lane & 15) == 0) {
        int rb = ibase + (lane >> 4) * 8;
#pragma unroll
        for (int r = 0; r < 8; ++r) {
            ms[b * NPix + rb + r] = m[r];
            ls[b * NPix + rb + r] = l[r];
        }
    }
}

// ---------------------------------------------------------------------------
// K4: flash pass 2.  Recompute S from async-staged K-panels (two 16-col
//     panels per 32-col group), P = exp(S-m)/l through LDS (layout swap),
//     O += P*V via WMMA (reuse-A across the 4 c-tiles).
// ---------------------------------------------------------------------------
__global__ void attn_out_kernel(const half_t* __restrict__ qT,
                                const half_t* __restrict__ kT,
                                const half_t* __restrict__ vM,
                                const float* __restrict__ ms,
                                const float* __restrict__ ls,
                                half_t* __restrict__ attnT) {
    int b = blockIdx.z;
    int cbase = blockIdx.y * 64;
    int wave  = threadIdx.x >> 5;
    int ibase = (blockIdx.x * 4 + wave) * 16;
    const half_t* qb = qT + (size_t)b * NPix * CDIM;
    const half_t* kb = kT + (size_t)b * NPix * CDIM;
    const half_t* vb = vM + (size_t)b * CDIM * NPix;

    __shared__ __align__(128) half_t panel[2][16 * CDIM];   // 2 x 16KB
    __shared__ __align__(128) half_t pt[4][16 * 32];        // P tile per wave
    uint32_t pbase0 = lds_addr_of(&panel[0][0]);
    uint32_t pbase1 = lds_addr_of(&panel[1][0]);
    half_t* myp = &pt[wave][0];

    v16h qf[16];
#pragma unroll
    for (int ks = 0; ks < 16; ++ks)
        qf[ks] = load_a_f16(qb + (size_t)ibase * CDIM + ks * 32, CDIM);

    int lane = threadIdx.x & 31;
    int hsel = lane >> 4, col = lane & 15;
    float mrow[8], linv[8];
#pragma unroll
    for (int r = 0; r < 8; ++r) {
        int row = ibase + r + hsel * 8;
        mrow[r] = ms[b * NPix + row];
        linv[r] = 1.0f / ls[b * NPix + row];
    }

    v8f o[4] = {zero8(), zero8(), zero8(), zero8()};
    v8f accE = zero8();                            // even-panel S columns

    stage_panel_async(pbase0, kb);                 // panel h=0
    for (int h = 0; h < NPix / 16; ++h) {
        const half_t* pp = (h & 1) ? &panel[1][0] : &panel[0][0];
        if (h + 1 < NPix / 16) {
            stage_panel_async((h & 1) ? pbase0 : pbase1,
                              kb + (size_t)(h + 1) * 16 * CDIM);
            S_WAIT_ASYNCCNT(8);
        } else {
            S_WAIT_ASYNCCNT(0);
        }
        __syncthreads();

        v8f sacc = zero8();
#pragma unroll
        for (int ks = 0; ks < 16; ++ks)
            sacc = wmma_f16(qf[ks], load_b_kcontig(pp + ks * 32, CDIM), sacc);

        if ((h & 1) == 0) {
            accE = sacc;                           // first half of 32-col group
        } else {
            int j0 = (h - 1) * 16;
            // P (16x32) into LDS: C-layout -> row-major tile
#pragma unroll
            for (int r = 0; r < 8; ++r) {
                int row = r + hsel * 8;
                myp[row * 32 + col]      = (half_t)(__expf(accE[r] - mrow[r]) * linv[r]);
                myp[row * 32 + 16 + col] = (half_t)(__expf(sacc[r] - mrow[r]) * linv[r]);
            }
            S_WAIT_DSCNT0();                       // cross-lane LDS hazard
            v16h pa  = load_a_f16(myp, 32);        // A-fragment of P
            v16h bv0 = load_b_kcontig(vb + (size_t)(cbase + 0)  * NPix + j0, NPix);
            v16h bv1 = load_b_kcontig(vb + (size_t)(cbase + 16) * NPix + j0, NPix);
            v16h bv2 = load_b_kcontig(vb + (size_t)(cbase + 32) * NPix + j0, NPix);
            v16h bv3 = load_b_kcontig(vb + (size_t)(cbase + 48) * NPix + j0, NPix);
            o[0] = wmma_f16(pa, bv0, o[0]);
            o[1] = wmma_f16_ra(pa, bv1, o[1]);
            o[2] = wmma_f16_ra(pa, bv2, o[2]);
            o[3] = wmma_f16_ra(pa, bv3, o[3]);
        }
        __syncthreads();                           // done reading this buffer
    }

#pragma unroll
    for (int t = 0; t < 4; ++t) {
#pragma unroll
        for (int r = 0; r < 8; ++r) {
            int row = ibase + r + hsel * 8;
            attnT[(size_t)b * NPix * CDIM + (size_t)row * CDIM + cbase + t * 16 + col] =
                (half_t)o[t][r];
        }
    }
}

// ---------------------------------------------------------------------------
// K5: output projection + residual:  y = x + Wp * attn + bp
// ---------------------------------------------------------------------------
__global__ void proj_out_kernel(const half_t* __restrict__ attnT,
                                const half_t* __restrict__ whp,
                                const float* __restrict__ bp,
                                const float* __restrict__ x,
                                float* __restrict__ out) {
    int b = blockIdx.z;
    int wave  = threadIdx.x >> 5;
    int dbase = (blockIdx.y * 4 + wave) * 16;
    int nbase = blockIdx.x * 64;
    const half_t* ab = attnT + (size_t)b * NPix * CDIM;

    v8f acc[4] = {zero8(), zero8(), zero8(), zero8()};
    gemm_row_panel(whp, dbase, ab, nbase, acc);

    int lane = threadIdx.x & 31;
    int hsel = lane >> 4, col = lane & 15;
    float bias_r[8];
#pragma unroll
    for (int r = 0; r < 8; ++r) bias_r[r] = bp[dbase + r + hsel * 8];

#pragma unroll
    for (int t = 0; t < 4; ++t) {
#pragma unroll
        for (int r = 0; r < 8; ++r) {
            int d = dbase + r + hsel * 8;
            int n = nbase + t * 16 + col;
            size_t idx = (size_t)b * CDIM * NPix + (size_t)d * NPix + n;
            out[idx] = x[idx] + acc[t][r] + bias_r[r];
        }
    }
}

// ---------------------------------------------------------------------------
extern "C" void kernel_launch(void* const* d_in, const int* in_sizes, int n_in,
                              void* d_out, int out_size, void* d_ws, size_t ws_size,
                              hipStream_t stream) {
    (void)in_sizes; (void)n_in; (void)out_size; (void)ws_size;
    const float* x   = (const float*)d_in[0];
    const float* gsc = (const float*)d_in[1];
    const float* gbs = (const float*)d_in[2];
    const float* wq  = (const float*)d_in[3];
    const float* bq  = (const float*)d_in[4];
    const float* wk  = (const float*)d_in[5];
    const float* bk  = (const float*)d_in[6];
    const float* wv  = (const float*)d_in[7];
    const float* bv  = (const float*)d_in[8];
    const float* wp  = (const float*)d_in[9];
    const float* bp  = (const float*)d_in[10];
    float* out = (float*)d_out;

    const size_t ELEMS = (size_t)BATCH * NPix * CDIM;
    const size_t SZH   = ELEMS * sizeof(half_t);          // 8 MB per f16 tensor
    char* ws = (char*)d_ws;
    half_t* hnT   = (half_t*)(ws);
    half_t* qT    = (half_t*)(ws + 1 * SZH);
    half_t* kT    = (half_t*)(ws + 2 * SZH);
    half_t* vM    = (half_t*)(ws + 3 * SZH);
    half_t* attnT = (half_t*)(ws + 4 * SZH);
    float*  ms    = (float*)(ws + 5 * SZH);
    float*  ls    = (float*)(ws + 5 * SZH + (size_t)BATCH * NPix * sizeof(float));
    half_t* wh    = (half_t*)(ws + 5 * SZH + 2 * (size_t)BATCH * NPix * sizeof(float));
    // wh layout: [q|k|v|p] each 512*512 f16 (2 MB total)

    wcvt_kernel<<<dim3(128, 4), 256, 0, stream>>>(wq, wk, wv, wp, wh);
    gn_kernel<<<dim3(BATCH * NGROUPS), 256, 0, stream>>>(x, gsc, gbs, hnT);
    qkv_kernel<<<dim3(NPix / 64, CDIM / 16 / 4, BATCH * 3), 128, 0, stream>>>(
        hnT, wh, bq, bk, bv, qT, kT, vM);
    stats_kernel<<<dim3(NPix / 16 / 4, BATCH), 128, 0, stream>>>(qT, kT, ms, ls);
    attn_out_kernel<<<dim3(NPix / 16 / 4, CDIM / 64, BATCH), 128, 0, stream>>>(
        qT, kT, vM, ms, ls, attnT);
    proj_out_kernel<<<dim3(NPix / 64, CDIM / 16 / 4, BATCH), 128, 0, stream>>>(
        attnT, wh + 3 * (size_t)WELEMS, bp, x, out);
}
